// GAT_30562987278371
// MI455X (gfx1250) — matrix operand order repaired and checked by reading, not compiled, not run on hardware
//
#include <hip/hip_runtime.h>

typedef __attribute__((ext_vector_type(16))) __bf16 v16bf;
typedef __attribute__((ext_vector_type(8)))  float  v8f;
typedef __attribute__((ext_vector_type(2)))  __bf16 bf16x2;

#define NEG_SLOPE 0.2f

// ---------------------------------------------------------------------------
// CSR construction (built once, reused by all 3 layers)
// ---------------------------------------------------------------------------
__global__ void count_kernel(const int* __restrict__ ei, int* __restrict__ deg,
                             int E, int EE) {
  int t = blockIdx.x * blockDim.x + threadIdx.x;
  if (t >= EE) return;
  int dst = (t < E) ? ei[E + t] : (t - E);   // self-loops appended
  atomicAdd(&deg[dst], 1);
}

__global__ void scan_kernel(const int* __restrict__ deg, int* __restrict__ row_ptr,
                            int* __restrict__ cursor, int n) {
  __shared__ int buf[1024];
  __shared__ int carry;
  const int t = threadIdx.x;
  if (t == 0) { carry = 0; row_ptr[0] = 0; }
  __syncthreads();
  for (int base = 0; base < n; base += 1024) {
    const int i = base + t;
    int v = (i < n) ? deg[i] : 0;
    buf[t] = v;
    __syncthreads();
    for (int off = 1; off < 1024; off <<= 1) {
      int tv = (t >= off) ? buf[t - off] : 0;
      __syncthreads();
      buf[t] += tv;
      __syncthreads();
    }
    if (i < n) {
      int incl = carry + buf[t];
      row_ptr[i + 1] = incl;
      cursor[i] = incl - v;          // exclusive prefix = write cursor
    }
    __syncthreads();
    if (t == 1023) carry += buf[1023];
    __syncthreads();
  }
}

__global__ void fill_kernel(const int* __restrict__ ei, int* __restrict__ cursor,
                            int* __restrict__ csr_src, int E, int EE) {
  int t = blockIdx.x * blockDim.x + threadIdx.x;
  if (t >= EE) return;
  int src, dst;
  if (t < E) { src = ei[t]; dst = ei[E + t]; }
  else       { src = dst = t - E; }
  int pos = atomicAdd(&cursor[dst], 1);
  csr_src[pos] = src;
}

// ---------------------------------------------------------------------------
// Weight transpose + f32->bf16 convert:  W[K,Fout] -> Wt[Fout][K] (bf16)
// Makes WMMA B-fragment K-pairs contiguous (merges into global_load_b128).
// ---------------------------------------------------------------------------
__global__ void wtrans_kernel(const float* __restrict__ W, __bf16* __restrict__ Wt,
                              int K, int Fout) {
  int t = blockIdx.x * blockDim.x + threadIdx.x;
  if (t >= K * Fout) return;
  int k = t / Fout, c = t % Fout;
  Wt[(size_t)c * K + k] = (__bf16)W[t];
}

// ---------------------------------------------------------------------------
// GEMM: out[N,FOUT] = in[N,FIN] @ W  via v_wmma_f32_16x16x32_bf16
// 128 threads = 4 waves; each wave owns CT adjacent 16x16 output tiles and
// reuses the A-fragment registers across the CT back-to-back WMMAs.
// A tile (16 x FIN) staged to LDS as bf16; A and B fragments both use contiguous k = 16*hlf + s order.
// ---------------------------------------------------------------------------
template<int FIN, int FOUT>
__global__ __launch_bounds__(128) void gemm_wmma_kernel(
    const float* __restrict__ in, const __bf16* __restrict__ Wt,
    float* __restrict__ out, int n_rows) {
  constexpr int CT  = (FOUT >= 128) ? 2 : 1;   // column tiles per wave
  constexpr int PAD = 8;
  __shared__ __bf16 Atile[16][FIN + PAD];
  const int tid = threadIdx.x;
  const int row_base = blockIdx.x * 16;
  const int col_base = (blockIdx.y * 4 + (tid >> 5)) * (16 * CT);

  // stage 16 x FIN activation tile -> bf16 LDS
  for (int i = tid; i < 16 * FIN; i += 128) {
    int r = i / FIN, k = i % FIN;
    int rr = row_base + r; if (rr >= n_rows) rr = n_rows - 1;
    Atile[r][k] = (__bf16)in[(size_t)rr * FIN + k];
  }
  __syncthreads();

  const int lane = tid & 31;
  const int hlf  = lane >> 4;     // lane group (0: lanes 0-15, 1: lanes 16-31)
  const int mrow = lane & 15;     // A row for this lane
  const int ncol = lane & 15;     // B/D column for this lane

  v8f acc[CT];
#pragma unroll
  for (int ct = 0; ct < CT; ++ct) acc[ct] = v8f{};

  for (int kk = 0; kk < FIN; kk += 32) {
    v16bf a;
#pragma unroll
    for (int i = 0; i < 8; ++i) {
      // A 16x32 bf16: vgpr i holds K = 16*hlf + 2i, +1
      int ka = kk + 16 * hlf + 2 * i;
      bf16x2 pa = *(const bf16x2*)&Atile[mrow][ka];
      a[2 * i]     = pa.x;
      a[2 * i + 1] = pa.y;
    }
#pragma unroll
    for (int ct = 0; ct < CT; ++ct) {
      const __bf16* wcol = Wt + (size_t)(col_base + 16 * ct + ncol) * FIN;
      v16bf b;
#pragma unroll
      for (int i = 0; i < 8; ++i) {
        // B 32x16 bf16: vgpr i holds K = 16*hlf + 2i, +1 (contiguous in Wt row)
        int kb = kk + 16 * hlf + 2 * i;
        bf16x2 pb = *(const bf16x2*)&wcol[kb];
        b[2 * i]     = pb.x;
        b[2 * i + 1] = pb.y;
      }
      acc[ct] = __builtin_amdgcn_wmma_f32_16x16x32_bf16(
          false, a, false, b, (short)0, acc[ct], false, false);
    }
  }

  // D 16x16 f32: vgpr v -> row = v + 8*hlf, col = lane%16
  if (row_base + 16 <= n_rows) {               // full tile: unguarded stores
#pragma unroll
    for (int ct = 0; ct < CT; ++ct) {
      float* op = out + (size_t)(row_base + 8 * hlf) * FOUT + col_base + 16 * ct + ncol;
#pragma unroll
      for (int v = 0; v < 8; ++v) op[(size_t)v * FOUT] = acc[ct][v];
    }
  } else {                                     // ragged last tile
#pragma unroll
    for (int ct = 0; ct < CT; ++ct)
#pragma unroll
      for (int v = 0; v < 8; ++v) {
        int r = row_base + v + 8 * hlf;
        if (r < n_rows)
          out[(size_t)r * FOUT + col_base + 16 * ct + ncol] = acc[ct][v];
      }
  }
}

// ---------------------------------------------------------------------------
// Per-node attention logits: alpha_s[n,h] = <h[n,h,:], a_src[h,:]>, same dst
// ---------------------------------------------------------------------------
template<int H, int C>
__global__ void alpha_kernel(const float* __restrict__ h,
                             const float* __restrict__ a_src,
                             const float* __restrict__ a_dst,
                             float* __restrict__ alpha_s,
                             float* __restrict__ alpha_d, int n_nodes) {
  int t = blockIdx.x * blockDim.x + threadIdx.x;
  if (t >= n_nodes * H) return;
  int n = t / H, hd = t % H;
  const float* hp = h + (size_t)n * (H * C) + hd * C;
  const float* as = a_src + hd * C;
  const float* ad = a_dst + hd * C;
  float ss = 0.f, sd = 0.f;
#pragma unroll
  for (int c = 0; c < C; c += 4) {
    float4 hv = *(const float4*)(hp + c);
    float4 av = *(const float4*)(as + c);
    float4 dv = *(const float4*)(ad + c);
    ss += hv.x * av.x + hv.y * av.y + hv.z * av.z + hv.w * av.w;
    sd += hv.x * dv.x + hv.y * dv.y + hv.z * dv.z + hv.w * dv.w;
  }
  alpha_s[t] = ss;
  alpha_d[t] = sd;
}

// ---------------------------------------------------------------------------
// One wave32 per destination node: online softmax over incoming edges
// (fused max+sum, per-head shfl_xor tree) then gather-accumulate of
// h[src] * alpha. Zero atomics; output (+bias, optional ELU) written once.
// ---------------------------------------------------------------------------
template<int H, int C, bool ELU>
__global__ __launch_bounds__(256) void gat_aggregate_kernel(
    const float* __restrict__ h,
    const float* __restrict__ alpha_s,
    const float* __restrict__ alpha_d,
    const int* __restrict__ row_ptr,
    const int* __restrict__ csr_src,
    const float* __restrict__ bias,
    float* __restrict__ out, int n_nodes) {
  constexpr int F   = H * C;
  constexpr int CH  = F / 32;   // channels per lane in pass 2
  constexpr int EPI = 32 / H;   // edges per iteration in pass 1
  const int lane = threadIdx.x & 31;
  const int n = blockIdx.x * (blockDim.x >> 5) + (threadIdx.x >> 5);
  if (n >= n_nodes) return;
  const int s = row_ptr[n], e = row_ptr[n + 1];

  // ---- pass 1: online softmax (running max m, rescaled sum ssum) ----
  const int hd1 = lane % H;
  const int eoff = lane / H;
  const float adn = alpha_d[n * H + hd1];
  float m = -3.4e38f, ssum = 0.f;    // finite sentinel: no inf-inf NaNs
  for (int i = s; i < e; i += EPI) {
    int idx = i + eoff;
    if (idx < e) {
      int sn = csr_src[idx];
      float v = alpha_s[sn * H + hd1] + adn;
      v = v > 0.f ? v : NEG_SLOPE * v;           // leaky_relu
      if (v > m) { ssum *= __expf(m - v); m = v; }
      ssum += __expf(v - m);
    }
  }
  for (int off = H; off < 32; off <<= 1) {       // per-head tree merge
    float mo = __shfl_xor(m, off, 32);
    float so = __shfl_xor(ssum, off, 32);
    float mn = fmaxf(m, mo);
    ssum = ssum * __expf(m - mn) + so * __expf(mo - mn);
    m = mn;
  }

  // ---- pass 2: gather-accumulate; lane owns CH contiguous channels ----
  const int c0 = lane * CH;
  const int hd3 = c0 / C;                        // head of this lane's channels
  const float m3  = __shfl(m, hd3, 32);          // lane hd3 holds head hd3
  const float d3  = __shfl(ssum, hd3, 32);
  const float ad3v = alpha_d[n * H + hd3];
  const float inv = 1.f / (d3 + 1e-16f);
  float acc[CH];
#pragma unroll
  for (int k = 0; k < CH; ++k) acc[k] = 0.f;
  for (int i = s; i < e; ++i) {
    int sn = csr_src[i];
    float t = alpha_s[sn * H + hd3] + ad3v;
    t = t > 0.f ? t : NEG_SLOPE * t;
    float al = __expf(t - m3) * inv;
    const float* hp = h + (size_t)sn * F + c0;
    if constexpr (CH == 2) {
      float2 hv = *(const float2*)hp;
      acc[0] += al * hv.x; acc[1] += al * hv.y;
    } else {
#pragma unroll
      for (int k = 0; k < CH; k += 4) {
        float4 hv = *(const float4*)(hp + k);
        acc[k + 0] += al * hv.x; acc[k + 1] += al * hv.y;
        acc[k + 2] += al * hv.z; acc[k + 3] += al * hv.w;
      }
    }
  }
#pragma unroll
  for (int k = 0; k < CH; ++k) {
    float v = acc[k] + bias[c0 + k];
    if (ELU) v = v > 0.f ? v : (__expf(v) - 1.f);
    out[(size_t)n * F + c0 + k] = v;
  }
}

// ---------------------------------------------------------------------------
extern "C" void kernel_launch(void* const* d_in, const int* in_sizes, int n_in,
                              void* d_out, int out_size, void* d_ws, size_t ws_size,
                              hipStream_t stream) {
  const float* x   = (const float*)d_in[0];
  const int*   ei  = (const int*)  d_in[1];
  const float* W1  = (const float*)d_in[2];
  const float* as1 = (const float*)d_in[3];
  const float* ad1 = (const float*)d_in[4];
  const float* b1  = (const float*)d_in[5];
  const float* W2  = (const float*)d_in[6];
  const float* as2 = (const float*)d_in[7];
  const float* ad2 = (const float*)d_in[8];
  const float* b2  = (const float*)d_in[9];
  const float* W3  = (const float*)d_in[10];
  const float* as3 = (const float*)d_in[11];
  const float* ad3 = (const float*)d_in[12];
  const float* b3  = (const float*)d_in[13];

  const int N_ = in_sizes[0] / 128;   // 20000
  const int E_ = in_sizes[1] / 2;     // 640000
  const int EE = E_ + N_;             // + self loops

  char* p = (char*)d_ws;
  auto take = [&](size_t bytes) -> void* {
    void* r = (void*)p;
    p += (bytes + 255) & ~(size_t)255;
    return r;
  };
  float*  hA      = (float*) take((size_t)N_ * 256 * 4);
  float*  hB      = (float*) take((size_t)N_ * 256 * 4);
  float*  aS      = (float*) take((size_t)N_ * 8 * 4);
  float*  aD      = (float*) take((size_t)N_ * 8 * 4);
  int*    deg     = (int*)   take((size_t)N_ * 4);
  int*    cursor  = (int*)   take((size_t)N_ * 4);
  int*    row_ptr = (int*)   take((size_t)(N_ + 1) * 4);
  int*    csr     = (int*)   take((size_t)EE * 4);
  __bf16* Wt1     = (__bf16*)take((size_t)128 * 256 * 2);
  __bf16* Wt2     = (__bf16*)take((size_t)256 * 256 * 2);
  __bf16* Wt3     = (__bf16*)take((size_t)256 * 64 * 2);

  // --- CSR build (once) ---
  hipMemsetAsync(deg, 0, (size_t)N_ * 4, stream);
  count_kernel<<<(EE + 255) / 256, 256, 0, stream>>>(ei, deg, E_, EE);
  scan_kernel<<<1, 1024, 0, stream>>>(deg, row_ptr, cursor, N_);
  fill_kernel<<<(EE + 255) / 256, 256, 0, stream>>>(ei, cursor, csr, E_, EE);

  // --- weight convert/transpose to bf16 (once) ---
  wtrans_kernel<<<(128 * 256 + 255) / 256, 256, 0, stream>>>(W1, Wt1, 128, 256);
  wtrans_kernel<<<(256 * 256 + 255) / 256, 256, 0, stream>>>(W2, Wt2, 256, 256);
  wtrans_kernel<<<(256 * 64 + 255) / 256, 256, 0, stream>>>(W3, Wt3, 256, 64);

  const int rt = (N_ + 15) / 16;   // row tiles (N=20000 -> exactly 1250)

  // --- layer 1: x[N,128] -> 8 heads x 32, ELU ---
  gemm_wmma_kernel<128, 256><<<dim3(rt, 2), 128, 0, stream>>>(x, Wt1, hA, N_);
  alpha_kernel<8, 32><<<(N_ * 8 + 255) / 256, 256, 0, stream>>>(hA, as1, ad1, aS, aD, N_);
  gat_aggregate_kernel<8, 32, true><<<(N_ + 7) / 8, 256, 0, stream>>>(
      hA, aS, aD, row_ptr, csr, b1, hB, N_);

  // --- layer 2: 256 -> 8 heads x 32, ELU ---
  gemm_wmma_kernel<256, 256><<<dim3(rt, 2), 128, 0, stream>>>(hB, Wt2, hA, N_);
  alpha_kernel<8, 32><<<(N_ * 8 + 255) / 256, 256, 0, stream>>>(hA, as2, ad2, aS, aD, N_);
  gat_aggregate_kernel<8, 32, true><<<(N_ + 7) / 8, 256, 0, stream>>>(
      hA, aS, aD, row_ptr, csr, b2, hB, N_);

  // --- layer 3: 256 -> 1 head x 64, no activation, mean(1 head)=identity ---
  gemm_wmma_kernel<256, 64><<<dim3(rt, 1), 128, 0, stream>>>(hB, Wt3, hA, N_);
  alpha_kernel<1, 64><<<(N_ + 255) / 256, 256, 0, stream>>>(hA, as3, ad3, aS, aD, N_);
  gat_aggregate_kernel<1, 64, false><<<(N_ + 7) / 8, 256, 0, stream>>>(
      hA, aS, aD, row_ptr, csr, b3, (float*)d_out, N_);
}
